// NUFNOBlock_68590627717183
// MI455X (gfx1250) — matrix operand order repaired and checked by reading, not compiled, not run on hardware
//
#include <hip/hip_runtime.h>
#include <hip/hip_bf16.h>
#include <hip/hip_fp16.h>

// NUFNO block for MI455X (gfx1250, wave32, WMMA f16->f32).
// Scratch layout (requires ws_size >= ~9.3 MB):
//   [0)          x as f16,          4*64*16384 halves   (8,388,608 B)
//   [8388608)    x_hat^T f32,       4*512*64 floats     (  524,288 B)
//   [8912896)    A2 f16 (folded),   4*64*576 halves     (  294,912 B)
//   [9207808)    bias2 f32,         4*64 floats         (    1,024 B)

typedef __attribute__((ext_vector_type(16))) _Float16 v16h;
typedef __attribute__((ext_vector_type(8)))  _Float16 v8h;
typedef __attribute__((ext_vector_type(8)))  float    v8f;

#define NB 4
#define NC 64
#define NP 16384
#define NMODE 256
#define N2 512            // 2*NMODE (cos/sin interleaved rows)
#define KD 576            // N2 + NC
#define NEMB 256
#define NHID 64
#define KSW 256           // forward split-K slice width (p)
#define TWO_PI 6.2831853071795864769f

#define WS_XF16  0
#define WS_XHAT  8388608
#define WS_A2    8912896
#define WS_BIAS2 9207808

// A-operand fragment (16-bit, 16x32): lane m=lane&15 is the row; halves
// e0..15 hold K = 16*hi + e
// (hi = lane>=16).  One contiguous 32B chunk per lane.
static __device__ inline v16h load_a_chunks(const _Float16* row, int hi) {
    return *(const v16h*)(row + 16 * hi);
}

static __device__ inline float silu_f(float v) {
    return v / (1.f + __expf(-v));
}

// ---------------- Kernel 1: x -> f16 copy, zero x_hat accumulator ----------
__global__ __launch_bounds__(256) void k_prep(const float* __restrict__ x,
                                              _Float16* __restrict__ xf16,
                                              float* __restrict__ xhat) {
    int idx = blockIdx.x * 256 + threadIdx.x;
    if (idx < NB * NC * NP) xf16[idx] = (_Float16)x[idx];
    if (idx < NB * N2 * NC) xhat[idx] = 0.f;
}

// ---------------- Kernel 2: forward NUFFT  x_hat^T[n][c] = sum_p E[n][p]*x[c][p]
// grid: (NP/KSW, N2/64, NB), block 128 (4 waves). Split-K with f32 atomics.
// Wave tiling 2x2 (2 E-row tiles x 2 c-tiles): each x B-frag shared by 2 waves.
__global__ __launch_bounds__(128) void k_forward(const _Float16* __restrict__ xf16,
                                                 const float* __restrict__ pos,
                                                 float* __restrict__ xhat) {
    __shared__ _Float16 Et[64 * KSW];  // 32 KB, [n_local][p_local]
    const int ks = blockIdx.x, mg = blockIdx.y, b = blockIdx.z;
    const int t = threadIdx.x, lane = t & 31, w = t >> 5;
    const int m = lane & 15, hi = (lane >> 4) & 1;
    const int mt0 = 2 * (w & 1);      // wave's E-row tile pair
    const int ct0 = 2 * (w >> 1);     // wave's c tile pair
    const int p0 = ks * KSW;

    // Phase 1: E rows for modes [mg*32, mg*32+32) over this p slice.
    // mode = k1i*16 + k2i ; k1i in {2mg, 2mg+1}; k values = idx - 8.
    for (int pl = t; pl < KSW; pl += 128) {
        int p = p0 + pl;
        float px = pos[(size_t)(b * NP + p) * 2 + 0];
        float py = pos[(size_t)(b * NP + p) * 2 + 1];
        float c1[2], s1[2];
        __sincosf(TWO_PI * (float)(2 * mg - 8) * px, &s1[0], &c1[0]);
        __sincosf(TWO_PI * (float)(2 * mg - 7) * px, &s1[1], &c1[1]);
#pragma unroll
        for (int j = 0; j < 16; ++j) {
            float s2, c2;
            __sincosf(TWO_PI * (float)(j - 8) * py, &s2, &c2);
#pragma unroll
            for (int h = 0; h < 2; ++h) {
                int mm = h * 16 + j;  // local mode
                Et[(2 * mm + 0) * KSW + pl] = (_Float16)(c1[h] * c2 - s1[h] * s2);
                Et[(2 * mm + 1) * KSW + pl] = (_Float16)(s1[h] * c2 + c1[h] * s2);
            }
        }
    }
    __syncthreads();

    // Phase 2: WMMA 2x2 per wave per k-step.
    v8f z = {0, 0, 0, 0, 0, 0, 0, 0};
    v8f acc[2][2];
#pragma unroll
    for (int i = 0; i < 2; ++i)
#pragma unroll
        for (int j = 0; j < 2; ++j) acc[i][j] = z;

    for (int kk = 0; kk < KSW / 32; ++kk) {
        v16h af[2], bf[2];
#pragma unroll
        for (int i = 0; i < 2; ++i)
            af[i] = load_a_chunks(&Et[((mt0 + i) * 16 + m) * KSW + kk * 32], hi);
#pragma unroll
        for (int j = 0; j < 2; ++j) {
            // B fragment from x^T: lane is column c, halves are K=p (e + 16*hi).
            const _Float16* brow = xf16 +
                (size_t)(b * NC + (ct0 + j) * 16 + m) * NP + p0 + kk * 32 + 16 * hi;
            bf[j] = *(const v16h*)brow;
        }
#pragma unroll
        for (int i = 0; i < 2; ++i)
#pragma unroll
            for (int j = 0; j < 2; ++j)
                acc[i][j] = __builtin_amdgcn_wmma_f32_16x16x32_f16(
                    false, af[i], false, bf[j], (short)0, acc[i][j], false, false);
    }

    // Epilogue: C/D layout — VGPR r holds row (r + 8*hi), lane&15 is column.
#pragma unroll
    for (int i = 0; i < 2; ++i)
#pragma unroll
        for (int j = 0; j < 2; ++j)
#pragma unroll
            for (int r = 0; r < 8; ++r) {
                int row = mg * 64 + (mt0 + i) * 16 + r + 8 * hi;
                int col = (ct0 + j) * 16 + m;
                atomicAdd(&xhat[(size_t)b * N2 * NC + row * NC + col], acc[i][j][r]);
            }
}

// ---------------- Kernel 3: spectral mix + modulation + MLP; build A2 -------
// grid: (NB), block 256.
__global__ __launch_bounds__(256) void k_mix(
    const float* __restrict__ emb, const float* __restrict__ wr,
    const float* __restrict__ wi, const float* __restrict__ modr,
    const float* __restrict__ modi, const float* __restrict__ W1,
    const float* __restrict__ b1, const float* __restrict__ W2,
    const float* __restrict__ b2, const float* __restrict__ wc,
    const float* __restrict__ bc, const float* __restrict__ xhat,
    _Float16* __restrict__ A2, float* __restrict__ bias2) {
    __shared__ float hbuf[NHID], gbuf[2 * NC], gam1[NC], phr[NMODE], phq[NMODE];
    const int b = blockIdx.x, t = threadIdx.x;
    const float* e = emb + b * NEMB;

    if (t < NHID) {  // hidden = silu(emb @ W1 + b1)
        float a = b1[t];
        for (int c = 0; c < NEMB; ++c) a += e[c] * W1[c * NHID + t];
        hbuf[t] = silu_f(a);
    }
    __syncthreads();
    if (t < 2 * NC) {  // h = hidden @ W2 + b2 -> gamma|beta
        float a = b2[t];
        for (int h = 0; h < NHID; ++h) a += hbuf[h] * W2[h * 2 * NC + t];
        gbuf[t] = a;
    }
    if (t < NMODE) {  // phi[mode] = sum_c mod[mode][c] * emb[c]
        float pr = 0.f, pi = 0.f;
        for (int c = 0; c < NEMB; ++c) {
            float ec = e[c];
            pr += modr[t * NEMB + c] * ec;
            pi += modi[t * NEMB + c] * ec;
        }
        phr[t] = pr;
        phq[t] = pi;
    }
    __syncthreads();
    if (t < NC) {
        float g1 = 1.f + gbuf[t];
        gam1[t] = g1;
        bias2[b * NC + t] = bc[t] * g1 + gbuf[NC + t];
    }
    __syncthreads();

    // A2[o][2*mode(+1)] = (2/P) * phi * (sum_i xhat[i][mode] * w[i][o][mode])
    const float sc = 2.f / (float)NP;
    for (int q = t; q < NC * NMODE; q += 256) {
        int o = q >> 8, md = q & 255;
        float accr = 0.f, acci = 0.f;
        for (int i = 0; i < NC; ++i) {
            float ar = xhat[(size_t)b * N2 * NC + (2 * md) * NC + i];
            float ai = xhat[(size_t)b * N2 * NC + (2 * md + 1) * NC + i];
            float wre = wr[(size_t)(i * NC + o) * NMODE + md];
            float wim = wi[(size_t)(i * NC + o) * NMODE + md];
            accr += ar * wre - ai * wim;
            acci += ar * wim + ai * wre;
        }
        float yr = phr[md] * accr - phq[md] * acci;
        float yi = phr[md] * acci + phq[md] * accr;
        A2[(size_t)(b * NC + o) * KD + 2 * md]     = (_Float16)(sc * yr);
        A2[(size_t)(b * NC + o) * KD + 2 * md + 1] = (_Float16)(sc * yi);
    }
    // A2[o][512+i] = wc[o][i] * (1+gamma[o])   (residual path folded in)
    for (int q = t; q < NC * NC; q += 256) {
        int o = q >> 6, i = q & 63;
        A2[(size_t)(b * NC + o) * KD + N2 + i] = (_Float16)(wc[o * NC + i] * gam1[o]);
    }
}

// ---------------- Kernel 4: inverse NUFFT + residual + SiLU -----------------
// out[c][p] = silu( sum_k A2[c][k]*B[k][p] + bias2[c] ),  K = 512 E-rows + 64 x-rows
// grid: (NP/128, NB), block 256 (8 waves): 4 m-tiles x 2 p-halves.
// E tile is double-buffered: sincos generation for step kk+1 overlaps the
// WMMAs of step kk; one barrier per k-step.
__global__ __launch_bounds__(256) void k_inverse(const float* __restrict__ x,
                                                 const float* __restrict__ pos,
                                                 const _Float16* __restrict__ A2,
                                                 const float* __restrict__ bias2,
                                                 float* __restrict__ out) {
    __shared__ _Float16 Et[2][128 * 32];  // ping-pong [p][n_local]  2x8 KB
    __shared__ _Float16 xT[128 * 64];     // [p][i]                 16 KB
    const int b = blockIdx.y;
    const int pbase = blockIdx.x * 128;
    const int t = threadIdx.x, lane = t & 31, w = t >> 5;
    const int m = lane & 15, hi = (lane >> 4) & 1;
    const int mt = w & 3, nh = w >> 2;

    float px = 0.f, c2[16], s2[16];
    if (t < 128) {  // cache k2 basis per owned p
        px = pos[(size_t)(b * NP + pbase + t) * 2 + 0];
        float py = pos[(size_t)(b * NP + pbase + t) * 2 + 1];
#pragma unroll
        for (int j = 0; j < 16; ++j)
            __sincosf(TWO_PI * (float)(j - 8) * py, &s2[j], &c2[j]);
        // prologue: E tile for kk = 0
        float s1, c1;
        __sincosf(TWO_PI * (float)(0 - 8) * px, &s1, &c1);
#pragma unroll
        for (int j = 0; j < 16; ++j) {
            Et[0][t * 32 + 2 * j]     = (_Float16)(c1 * c2[j] - s1 * s2[j]);
            Et[0][t * 32 + 2 * j + 1] = (_Float16)(s1 * c2[j] + c1 * s2[j]);
        }
    }
    __syncthreads();

    v8f z = {0, 0, 0, 0, 0, 0, 0, 0};
    v8f acc[4];
#pragma unroll
    for (int j = 0; j < 4; ++j) acc[j] = z;

    for (int kk = 0; kk < 18; ++kk) {
        // Produce operands for step kk+1 into the other buffer (no conflict
        // with this step's reads; barrier at loop end publishes them).
        int nxt = kk + 1;
        if (nxt < 16) {
            if (t < 128) {
                float s1, c1;
                __sincosf(TWO_PI * (float)(nxt - 8) * px, &s1, &c1);
                _Float16* eb = &Et[nxt & 1][t * 32];
#pragma unroll
                for (int j = 0; j < 16; ++j) {
                    eb[2 * j]     = (_Float16)(c1 * c2[j] - s1 * s2[j]);
                    eb[2 * j + 1] = (_Float16)(s1 * c2[j] + c1 * s2[j]);
                }
            }
        } else if (nxt == 16) {  // stage x^T tile once (serves steps 16,17)
            for (int q = t; q < 128 * 64; q += 256) {
                int i = q >> 7, pc = q & 127;
                xT[pc * 64 + i] = (_Float16)x[(size_t)(b * NC + i) * NP + pbase + pc];
            }
        }

        // Consume step kk.
        v16h af = load_a_chunks(&A2[(size_t)(b * NC + mt * 16 + m) * KD + kk * 32], hi);
#pragma unroll
        for (int j = 0; j < 4; ++j) {
            int pl = nh * 64 + j * 16 + m;  // lane's column p
            const _Float16* bp = (kk < 16)
                                     ? &Et[kk & 1][pl * 32 + 16 * hi]
                                     : &xT[pl * 64 + (kk - 16) * 32 + 16 * hi];
            v16h bf = *(const v16h*)bp;
            acc[j] = __builtin_amdgcn_wmma_f32_16x16x32_f16(
                false, af, false, bf, (short)0, acc[j], false, false);
        }
        __syncthreads();
    }

#pragma unroll
    for (int j = 0; j < 4; ++j)
#pragma unroll
        for (int r = 0; r < 8; ++r) {
            int c = mt * 16 + r + 8 * hi;
            int p = pbase + nh * 64 + j * 16 + m;
            float v = acc[j][r] + bias2[b * NC + c];
            out[(size_t)(b * NC + c) * NP + p] = silu_f(v);
        }
}

extern "C" void kernel_launch(void* const* d_in, const int* in_sizes, int n_in,
                              void* d_out, int out_size, void* d_ws, size_t ws_size,
                              hipStream_t stream) {
    const float* x    = (const float*)d_in[0];
    const float* pos  = (const float*)d_in[1];
    const float* emb  = (const float*)d_in[2];
    const float* wr   = (const float*)d_in[3];
    const float* wi   = (const float*)d_in[4];
    const float* modr = (const float*)d_in[5];
    const float* modi = (const float*)d_in[6];
    const float* W1   = (const float*)d_in[7];
    const float* b1   = (const float*)d_in[8];
    const float* W2   = (const float*)d_in[9];
    const float* b2   = (const float*)d_in[10];
    const float* wc   = (const float*)d_in[11];
    const float* bc   = (const float*)d_in[12];
    float* out = (float*)d_out;

    char* ws = (char*)d_ws;
    _Float16* xf16 = (_Float16*)(ws + WS_XF16);
    float* xhat    = (float*)(ws + WS_XHAT);
    _Float16* A2   = (_Float16*)(ws + WS_A2);
    float* bias2   = (float*)(ws + WS_BIAS2);

    k_prep<<<dim3((NB * NC * NP) / 256), 256, 0, stream>>>(x, xf16, xhat);
    k_forward<<<dim3(NP / KSW, N2 / 64, NB), 128, 0, stream>>>(xf16, pos, xhat);
    k_mix<<<dim3(NB), 256, 0, stream>>>(emb, wr, wi, modr, modi, W1, b1, W2, b2,
                                        wc, bc, xhat, A2, bias2);
    k_inverse<<<dim3(NP / 128, NB), 256, 0, stream>>>(x, pos, A2, bias2, out);
}
